// multi_head_attn_32049045963008
// MI455X (gfx1250) — matrix operand order repaired and checked by reading, not compiled, not run on hardware
//
#include <hip/hip_runtime.h>
#include <hip/hip_bf16.h>

// ---------------------------------------------------------------------------
// MI455X (gfx1250) windowed multi-head attention, wave32 + WMMA bf16.
// All GEMM stages use v_wmma_f32_16x16x32_bf16 (f32 accumulation).
// The residual stream is kept in f32 (renorm math) plus a bf16 mirror (xb)
// that feeds WMMA A-operands directly, so no f32->bf16 VALU in GEMM loops.
// ---------------------------------------------------------------------------

typedef __attribute__((ext_vector_type(16))) __bf16 v16bf;
typedef __attribute__((ext_vector_type(8)))  float  v8f;

constexpr int kS   = 2048;
constexpr int kD   = 512;
constexpr int kDH  = 64;
constexpr int kCW  = 128;
constexpr int kH   = 8;
constexpr int kVOFF  = 128;               // guard band before column 0 of vT
constexpr int kVPAD  = kS + 2 * kCW + 64; // 2368 padded columns in vT
constexpr int kWCOLS = 2 * kCW + 16;      // 272 key columns per query tile
constexpr int kWSTR  = 288;               // padded stride (multiple of 32)

union BF16x16 { v16bf v; unsigned short u[16]; uint4 q2[2]; };

__device__ __forceinline__ unsigned short f2bf(float f) {
  union { float f; unsigned u; } x; x.f = f;
  unsigned r = x.u + 0x7FFFu + ((x.u >> 16) & 1u);   // round-to-nearest-even
  return (unsigned short)(r >> 16);
}

// B fragment: lane = N column; 16 consecutive K values -> 32 contiguous bytes.
__device__ __forceinline__ v16bf load_bB(const unsigned short* p) {
  BF16x16 b;
  b.q2[0] = *(const uint4*)(p);
  b.q2[1] = *(const uint4*)(p + 8);
  return b.v;
}

// A fragment from bf16 row: 16 contiguous elems at c0 + (c0 & 8) (k = 16*hi + s).
__device__ __forceinline__ v16bf load_aB(const unsigned short* rowp, int c0) {
  BF16x16 a;
  a.q2[0] = *(const uint4*)(rowp + c0 + (c0 & 8));
  a.q2[1] = *(const uint4*)(rowp + c0 + (c0 & 8) + 8);
  return a.v;
}

__device__ __forceinline__ v8f wmma_bf16(v16bf a, v16bf b, v8f c) {
  return __builtin_amdgcn_wmma_f32_16x16x32_bf16(false, a, false, b,
                                                 (short)0, c, false, false);
}

__device__ __forceinline__ float wred_sum(float v) {
#pragma unroll
  for (int o = 16; o > 0; o >>= 1) v += __shfl_xor(v, o, 32);
  return v;
}
__device__ __forceinline__ float wred_max(float v) {
#pragma unroll
  for (int o = 16; o > 0; o >>= 1) v = fmaxf(v, __shfl_xor(v, o, 32));
  return v;
}

// ---------------------------------------------------------------------------
__global__ __launch_bounds__(256)
void cvt_f32_to_bf16(const float* __restrict__ src,
                     unsigned short* __restrict__ dst, int n) {
  int i = blockIdx.x * 256 + threadIdx.x;
  if (i < n) dst[i] = f2bf(src[i]);
}

// ---------------------------------------------------------------------------
// Q/K/V projection: one wave per 16-row tile; 12 accumulators (q,k,v x 4).
// xb: [S][512] bf16 mirror of the residual stream (A operand, no cvt needed).
// wq/wk/wv: [64][512] bf16 (row h contiguous in d) == B^T contiguous.
// q,k out: [S][64] bf16.  v out: transposed [64][VPAD] bf16 at col VOFF+s.
__global__ __launch_bounds__(256)
void proj_kernel(const unsigned short* __restrict__ xb,
                 const unsigned short* __restrict__ wq,
                 const unsigned short* __restrict__ wk,
                 const unsigned short* __restrict__ wv,
                 unsigned short* __restrict__ q,
                 unsigned short* __restrict__ k,
                 unsigned short* __restrict__ vT) {
  const int lane = threadIdx.x & 31;
  const int wave = threadIdx.x >> 5;
  const int tile = blockIdx.x * 8 + wave;
  const int j0   = tile * 16;
  const int hi   = lane >> 4;
  const int l16  = lane & 15;

  v8f zero = {0.f, 0.f, 0.f, 0.f, 0.f, 0.f, 0.f, 0.f};
  v8f acc[12];
#pragma unroll
  for (int i = 0; i < 12; ++i) acc[i] = zero;

  const unsigned short* arow = xb + (size_t)(j0 + l16) * kD;
  for (int kc = 0; kc < 16; ++kc) {
    v16bf a = load_aB(arow, kc * 32 + hi * 8);
#pragma unroll
    for (int nt = 0; nt < 4; ++nt) {
      const int woff = (nt * 16 + l16) * kD + kc * 32 + hi * 16;
      acc[nt]     = wmma_bf16(a, load_bB(wq + woff), acc[nt]);
      acc[4 + nt] = wmma_bf16(a, load_bB(wk + woff), acc[4 + nt]);
      acc[8 + nt] = wmma_bf16(a, load_bB(wv + woff), acc[8 + nt]);
    }
  }

  // q, k: C-layout (lane = column h, 8 rows) -> strided b16 stores.
#pragma unroll
  for (int nt = 0; nt < 4; ++nt) {
    const int h = nt * 16 + l16;
#pragma unroll
    for (int r = 0; r < 8; ++r) {
      const int m = r + hi * 8;
      q[(size_t)(j0 + m) * kDH + h] = f2bf(acc[nt][r]);
      k[(size_t)(j0 + m) * kDH + h] = f2bf(acc[4 + nt][r]);
    }
  }
  // v: 8 consecutive rows along s -> one packed 16B store per tile per lane.
#pragma unroll
  for (int nt = 0; nt < 4; ++nt) {
    const int h = nt * 16 + l16;
    union { uint4 q4; unsigned short u[8]; } pk;
#pragma unroll
    for (int r = 0; r < 8; ++r) pk.u[r] = f2bf(acc[8 + nt][r]);
    *(uint4*)(vT + (size_t)h * kVPAD + kVOFF + j0 + hi * 8) = pk.q4;
  }
}

// ---------------------------------------------------------------------------
// Windowed attention + up-projection + renorm + residual update.
// One wave per 16-query tile.  Writes cur_next (f32) and its bf16 mirror.
__global__ __launch_bounds__(32)
void attn_kernel(const float* __restrict__ cur,
                 const unsigned short* __restrict__ q,
                 const unsigned short* __restrict__ k,
                 const unsigned short* __restrict__ vT,
                 const unsigned short* __restrict__ wvu,
                 float* __restrict__ dst,
                 unsigned short* __restrict__ dstb) {
  __shared__ float          sy[16 * kD];    // scores (stride 288), then y (stride 512)
  __shared__ unsigned short wb[16 * kWSTR]; // softmax weights, bf16
  __shared__ unsigned short ab[16 * kDH];   // delta, bf16 (A staging)

  const int lane = threadIdx.x & 31;
  const int j0   = blockIdx.x * 16;
  const int hi   = lane >> 4;
  const int l16  = lane & 15;

  __builtin_prefetch(wvu + lane * kDH, 0, 3);   // global_prefetch_b8

  v8f zero = {0.f, 0.f, 0.f, 0.f, 0.f, 0.f, 0.f, 0.f};

  // ---- Phase 1: scores = q . k^T / sqrt(DH), masked to the +/-CW window ----
  const unsigned short* qrow = q + (size_t)(j0 + l16) * kDH;
  v16bf qa0 = load_aB(qrow, hi * 8);
  v16bf qa1 = load_aB(qrow, 32 + hi * 8);
  for (int kt = 0; kt < 17; ++kt) {
    const int key = j0 - kCW + kt * 16 + l16;
    const int kcl = key < 0 ? 0 : (key >= kS ? kS - 1 : key);
    v8f c = zero;
    c = wmma_bf16(qa0, load_bB(k + (size_t)kcl * kDH + hi * 16), c);
    c = wmma_bf16(qa1, load_bB(k + (size_t)kcl * kDH + 32 + hi * 16), c);
    const int w = kt * 16 + l16;
#pragma unroll
    for (int r = 0; r < 8; ++r) {
      const int m = r + hi * 8;
      const int d = w - m;
      const bool valid = (key >= 0) & (key < kS) & (d >= 0) & (d < 2 * kCW);
      sy[m * kWSTR + w] = valid ? c[r] * 0.125f : -__builtin_inff();
    }
  }
  __syncthreads();

  // ---- Phase 2: f32 softmax per row; weights to LDS as bf16 (cols 272..287=0)
  for (int m = 0; m < 16; ++m) {
    float mx = -__builtin_inff();
    for (int w = lane; w < kWCOLS; w += 32) mx = fmaxf(mx, sy[m * kWSTR + w]);
    mx = wred_max(mx);
    float sum = 0.f;
    for (int w = lane; w < kWCOLS; w += 32) {
      const float e = __expf(sy[m * kWSTR + w] - mx);
      sy[m * kWSTR + w] = e;           // cache exp, avoid recompute
      sum += e;
    }
    sum = wred_sum(sum);
    const float inv = 1.0f / sum;
    for (int w = lane; w < kWCOLS; w += 32)
      wb[m * kWSTR + w] = f2bf(sy[m * kWSTR + w] * inv);
    if (lane < 16) wb[m * kWSTR + kWCOLS + lane] = 0;
  }
  __syncthreads();

  // ---- Phase 3: delta = w @ v_win  (contraction over 288 padded keys) ------
  v8f dacc[4];
#pragma unroll
  for (int i = 0; i < 4; ++i) dacc[i] = zero;
  for (int kc = 0; kc < 9; ++kc) {
    v16bf a = load_aB(wb + l16 * kWSTR, kc * 32 + hi * 8);
#pragma unroll
    for (int nt = 0; nt < 4; ++nt) {
      const int dh = nt * 16 + l16;
      // vT column = VOFF + (j0 - CW) + kc*32 + hi*16 = j0 + kc*32 + hi*16
      v16bf b = load_bB(vT + (size_t)dh * kVPAD + j0 + kc * 32 + hi * 16);
      dacc[nt] = wmma_bf16(a, b, dacc[nt]);
    }
  }
#pragma unroll
  for (int nt = 0; nt < 4; ++nt)
#pragma unroll
    for (int r = 0; r < 8; ++r)
      ab[(r + hi * 8) * kDH + nt * 16 + l16] = f2bf(dacc[nt][r]);
  __syncthreads();

  // ---- Phase 4: y_delta = delta @ Wvu^T  (wvu stored [512][64] bf16) -------
  v16bf da0 = load_aB(ab + l16 * kDH, hi * 8);
  v16bf da1 = load_aB(ab + l16 * kDH, 32 + hi * 8);
  for (int nt = 0; nt < 32; ++nt) {
    const int drow = nt * 16 + l16;
    v8f c = zero;
    c = wmma_bf16(da0, load_bB(wvu + (size_t)drow * kDH + hi * 16), c);
    c = wmma_bf16(da1, load_bB(wvu + (size_t)drow * kDH + 32 + hi * 16), c);
#pragma unroll
    for (int r = 0; r < 8; ++r)
      sy[(r + hi * 8) * kD + nt * 16 + l16] = c[r];
  }
  __syncthreads();

  // ---- Phase 5: y = cur + y_delta; fused renorm; cur_next = cur + out ------
  for (int m = 0; m < 16; ++m) {
    const int row = j0 + m;
    const float* cp = cur + (size_t)row * kD + lane * 16;
    float cu[16], yv[16], ov[16];
    float s = 0.f;
#pragma unroll
    for (int i = 0; i < 4; ++i) {
      float4 c4 = *(const float4*)(cp + i * 4);
      float4 y4 = *(const float4*)(&sy[m * kD + lane * 16 + i * 4]);
      cu[i * 4 + 0] = c4.x; cu[i * 4 + 1] = c4.y;
      cu[i * 4 + 2] = c4.z; cu[i * 4 + 3] = c4.w;
      yv[i * 4 + 0] = c4.x + y4.x; yv[i * 4 + 1] = c4.y + y4.y;
      yv[i * 4 + 2] = c4.z + y4.z; yv[i * 4 + 3] = c4.w + y4.w;
      s += yv[i * 4 + 0] + yv[i * 4 + 1] + yv[i * 4 + 2] + yv[i * 4 + 3];
    }
    const float m1 = wred_sum(s) * (1.0f / kD);
    const float r1 = 1.0f / m1;
    float s1 = 0.f, s2 = 0.f;
#pragma unroll
    for (int i = 0; i < 16; ++i) {
      yv[i] *= r1;
      s1 += yv[i];
      s2 += yv[i] * yv[i];
    }
    s1 = wred_sum(s1);
    s2 = wred_sum(s2);
    const float m2  = s1 * (1.0f / kD);
    const float var = (s2 - (float)kD * m2 * m2) * (1.0f / (kD - 1));
    const float isd = rsqrtf(var);
#pragma unroll
    for (int i = 0; i < 16; ++i)
      ov[i] = cu[i] + (yv[i] - m2) * isd + m2;

    float* dp = dst + (size_t)row * kD + lane * 16;
#pragma unroll
    for (int i = 0; i < 4; ++i) {
      float4 o4;
      o4.x = ov[i * 4 + 0]; o4.y = ov[i * 4 + 1];
      o4.z = ov[i * 4 + 2]; o4.w = ov[i * 4 + 3];
      *(float4*)(dp + i * 4) = o4;
    }
    // bf16 mirror of the residual stream (A operand for next projection)
    unsigned short* bp = dstb + (size_t)row * kD + lane * 16;
#pragma unroll
    for (int i = 0; i < 2; ++i) {
      union { uint4 q4; unsigned short u[8]; } pk;
#pragma unroll
      for (int r = 0; r < 8; ++r) pk.u[r] = f2bf(ov[i * 8 + r]);
      *(uint4*)(bp + i * 8) = pk.q4;
    }
  }
}

// ---------------------------------------------------------------------------
__global__ __launch_bounds__(256)
void finalize_kernel(const float* __restrict__ src, float* __restrict__ dst, int n) {
  int i = blockIdx.x * 256 + threadIdx.x;
  if (i < n) dst[i] = src[i] * (1.0f / kH);
}

// ---------------------------------------------------------------------------
extern "C" void kernel_launch(void* const* d_in, const int* in_sizes, int n_in,
                              void* d_out, int out_size, void* d_ws, size_t ws_size,
                              hipStream_t stream) {
  (void)in_sizes; (void)n_in; (void)out_size; (void)ws_size;
  const float* x   = (const float*)d_in[0];
  const float* Wq  = (const float*)d_in[1];
  const float* Wk  = (const float*)d_in[2];
  const float* Wvd = (const float*)d_in[3];
  const float* Wvu = (const float*)d_in[4];
  float* out = (float*)d_out;

  char* ws = (char*)d_ws;
  size_t off = 0;
  auto take = [&](size_t bytes) -> char* {
    off = (off + 255) & ~(size_t)255;
    char* p = ws + off;
    off += bytes;
    return p;
  };
  float* curA = (float*)take((size_t)kS * kD * 4);
  float* curB = (float*)take((size_t)kS * kD * 4);
  unsigned short* xb   = (unsigned short*)take((size_t)kS * kD * 2);   // bf16 mirror
  unsigned short* qb   = (unsigned short*)take((size_t)kS * kDH * 2);
  unsigned short* kb   = (unsigned short*)take((size_t)kS * kDH * 2);
  unsigned short* vTb  = (unsigned short*)take((size_t)kDH * kVPAD * 2);
  unsigned short* wqb  = (unsigned short*)take((size_t)kH * kDH * kD * 2);
  unsigned short* wkb  = (unsigned short*)take((size_t)kH * kDH * kD * 2);
  unsigned short* wvdb = (unsigned short*)take((size_t)kH * kDH * kD * 2);
  unsigned short* wvub = (unsigned short*)take((size_t)kH * kD * kDH * 2);

  const int NW = kH * kDH * kD;  // 262144 elements per weight tensor
  cvt_f32_to_bf16<<<NW / 256, 256, 0, stream>>>(Wq,  wqb,  NW);
  cvt_f32_to_bf16<<<NW / 256, 256, 0, stream>>>(Wk,  wkb,  NW);
  cvt_f32_to_bf16<<<NW / 256, 256, 0, stream>>>(Wvd, wvdb, NW);
  cvt_f32_to_bf16<<<NW / 256, 256, 0, stream>>>(Wvu, wvub, NW);
  cvt_f32_to_bf16<<<(kS * kD) / 256, 256, 0, stream>>>(x, xb, kS * kD);

  const float* src = x;
  for (int h = 0; h < kH; ++h) {
    float* dstc = (h & 1) ? curB : curA;
    proj_kernel<<<(kS / 16) / 8, 256, 0, stream>>>(
        xb, wqb + (size_t)h * kDH * kD, wkb + (size_t)h * kDH * kD,
        wvdb + (size_t)h * kDH * kD, qb, kb, vTb);
    attn_kernel<<<kS / 16, 32, 0, stream>>>(
        src, qb, kb, vTb, wvub + (size_t)h * kD * kDH, dstc, xb);
    src = dstc;
  }
  finalize_kernel<<<(kS * kD) / 256, 256, 0, stream>>>(src, out, kS * kD);
}
